// DilatedNeuralNet_69114613730362
// MI455X (gfx1250) — matrix operand order repaired and checked by reading, not compiled, not run on hardware
//
#include <hip/hip_runtime.h>

// ---------------------------------------------------------------------------
// Dilated RNN (T=2048, H=1024, D1=2, D2=4) as a single persistent wave32
// workgroup on one WGP. Matvecs via V_WMMA_F32_16X16X32_BF16, vector in
// column 0 of the B fragment. Weights are pre-packed into a contiguous-k 16-bit
// A-fragment layout so each lane loads its fragment as 32 contiguous bytes.
// B fragments are loaded unconditionally: non-column-0 lanes read from a
// zero-filled LDS vector (no EXEC manipulation, no zero-fill movs).
// Requires ws_size >= 13 * 1024 * 1024 * 2 bytes (= 26 MB) for bf16 weights.
// ---------------------------------------------------------------------------

#define TT    2048
#define Hh    1024
#define D1c   2
#define D2c   4
#define KT    32          // k-tiles per matvec (1024 / 32)
#define TILE_E 512        // elements per packed 16x32 A tile

typedef __attribute__((ext_vector_type(16))) __bf16 bf16x16;
typedef __attribute__((ext_vector_type(8)))  float  f32x8;

__device__ __forceinline__ f32x8 wmma_bf16(bf16x16 a, bf16x16 b, f32x8 c) {
  // (neg_a, A, neg_b, B, c_mod, C, reuse_a, reuse_b)
  return __builtin_amdgcn_wmma_f32_16x16x32_bf16(false, a, false, b, (short)0, c,
                                                 false, false);
}

// y[0..7] = a + b, written as two ds_store_b128
__device__ __forceinline__ void store8(float* p, f32x8 a, f32x8 b) {
  float4 lo = make_float4(a[0] + b[0], a[1] + b[1], a[2] + b[2], a[3] + b[3]);
  float4 hi = make_float4(a[4] + b[4], a[5] + b[5], a[6] + b[6], a[7] + b[7]);
  *(float4*)p       = lo;
  *((float4*)p + 1) = hi;
}

// ---------------------------------------------------------------------------
// Pack row-major f32 HxH matrices into bf16 WMMA A-fragment order:
// tile (mt,kt) -> 512 bf16, lane l holds 16 contiguous bf16 covering
//   M = mt*16 + (l&15)
//   K = kt*32 + klocal,  klocal = ((l>>4)<<4) + e,  e = 0..15
// contiguous k order (not the ISA table order); B loads use the same order.
// ---------------------------------------------------------------------------
__global__ void pack_w(const float* __restrict__ W, __bf16* __restrict__ out,
                       int total) {
  int g = blockIdx.x * blockDim.x + threadIdx.x;
  if (g >= total) return;
  int mat  = g >> 20;               // / (H*H)
  int o    = g & (Hh * Hh - 1);
  int tile = o >> 9;                // / 512
  int r    = o & 511;
  int l    = r >> 4;                // lane 0..31
  int e    = r & 15;                // element 0..15
  int mt   = tile >> 5;
  int kt   = tile & 31;
  int m = mt * 16 + (l & 15);
  int k = kt * 32 + (((l >> 4) << 4) + e);
  out[g] = (__bf16)W[(size_t)mat * Hh * Hh + (size_t)m * Hh + k];
}

// y = A @ x : kt outer, both row tiles inner; 4 accumulator chains
// (2 row tiles x even/odd k) of depth 16 each.
__device__ __forceinline__ void matvec1_wmma(const __bf16* __restrict__ A,
                                             const __bf16* xsrc, float* yv,
                                             int lane, int wv, bool col0,
                                             int hlf) {
  const __bf16* pa0 = A + (size_t)(2 * wv + 0) * (KT * TILE_E) + lane * 16;
  const __bf16* pa1 = A + (size_t)(2 * wv + 1) * (KT * TILE_E) + lane * 16;
  f32x8 c00 = {}, c01 = {}, c10 = {}, c11 = {};
#pragma unroll 2
  for (int kt = 0; kt < KT; kt += 2) {
    bf16x16 bx0 = *(const bf16x16*)(xsrc + kt * 32 + hlf * 16);
    bf16x16 bx1 = *(const bf16x16*)(xsrc + (kt + 1) * 32 + hlf * 16);
    bf16x16 a00 = *(const bf16x16*)(pa0 + (size_t)kt * TILE_E);
    bf16x16 a01 = *(const bf16x16*)(pa0 + (size_t)(kt + 1) * TILE_E);
    bf16x16 a10 = *(const bf16x16*)(pa1 + (size_t)kt * TILE_E);
    bf16x16 a11 = *(const bf16x16*)(pa1 + (size_t)(kt + 1) * TILE_E);
    c00 = wmma_bf16(a00, bx0, c00);
    c01 = wmma_bf16(a01, bx1, c01);
    c10 = wmma_bf16(a10, bx0, c10);
    c11 = wmma_bf16(a11, bx1, c11);
  }
  if (col0) {
    store8(yv + (2 * wv + 0) * 16 + hlf * 8, c00, c01);
    store8(yv + (2 * wv + 1) * 16 + hlf * 8, c10, c11);
  }
}

// y = A @ x + Bm @ z : kt outer; B fragments shared by both row tiles;
// 4 accumulator chains (2 row tiles x 2 matrices) of depth 32 each.
__device__ __forceinline__ void matvec2_wmma(const __bf16* __restrict__ A,
                                             const __bf16* __restrict__ Bm,
                                             const __bf16* xsrc,
                                             const __bf16* zsrc, float* yv,
                                             int lane, int wv, bool col0,
                                             int hlf) {
  const __bf16* pa0 = A  + (size_t)(2 * wv + 0) * (KT * TILE_E) + lane * 16;
  const __bf16* pa1 = A  + (size_t)(2 * wv + 1) * (KT * TILE_E) + lane * 16;
  const __bf16* pb0 = Bm + (size_t)(2 * wv + 0) * (KT * TILE_E) + lane * 16;
  const __bf16* pb1 = Bm + (size_t)(2 * wv + 1) * (KT * TILE_E) + lane * 16;
  f32x8 cx0 = {}, cx1 = {}, cz0 = {}, cz1 = {};
#pragma unroll 2
  for (int kt = 0; kt < KT; ++kt) {
    bf16x16 bx = *(const bf16x16*)(xsrc + kt * 32 + hlf * 16);
    bf16x16 bz = *(const bf16x16*)(zsrc + kt * 32 + hlf * 16);
    bf16x16 a00 = *(const bf16x16*)(pa0 + (size_t)kt * TILE_E);
    bf16x16 a01 = *(const bf16x16*)(pa1 + (size_t)kt * TILE_E);
    bf16x16 a10 = *(const bf16x16*)(pb0 + (size_t)kt * TILE_E);
    bf16x16 a11 = *(const bf16x16*)(pb1 + (size_t)kt * TILE_E);
    cx0 = wmma_bf16(a00, bx, cx0);
    cx1 = wmma_bf16(a01, bx, cx1);
    cz0 = wmma_bf16(a10, bz, cz0);
    cz1 = wmma_bf16(a11, bz, cz1);
  }
  if (col0) {
    store8(yv + (2 * wv + 0) * 16 + hlf * 8, cx0, cz0);
    store8(yv + (2 * wv + 1) * 16 + hlf * 8, cx1, cz1);
  }
}

__global__ __launch_bounds__(1024, 1) void rnn_persistent(
    const float* __restrict__ q,
    const float* __restrict__ wih0, const float* __restrict__ bih0,
    const float* __restrict__ bhh0,
    const float* __restrict__ bih1, const float* __restrict__ bhh1,
    const float* __restrict__ bih2, const float* __restrict__ bhh2,
    const float* __restrict__ wlin, const float* __restrict__ blin,
    const __bf16* __restrict__ whh0p,
    const __bf16* __restrict__ wih1p, const __bf16* __restrict__ whh1p,
    const __bf16* __restrict__ wih2p, const __bf16* __restrict__ whh2p,
    float* __restrict__ out) {
  __shared__ float h0[Hh];
  __shared__ float h1[D1c][Hh];
  __shared__ float h2[D2c][Hh];
  __shared__ __attribute__((aligned(32))) float yv[Hh];
  __shared__ __attribute__((aligned(32))) __bf16 xb[Hh];
  __shared__ __attribute__((aligned(32))) __bf16 zb[Hh];
  __shared__ __attribute__((aligned(32))) __bf16 zerov[Hh];  // stays all-zero
  __shared__ float bsum0[Hh];
  __shared__ float bsum1[D1c][Hh];
  __shared__ float bsum2[D2c][Hh];
  __shared__ float wi0[Hh];
  __shared__ float wlo[Hh];
  __shared__ float predAcc;

  const int  t    = threadIdx.x;    // 0..1023
  const int  lane = t & 31;
  const int  wv   = t >> 5;         // wave 0..31, owns row tiles 2wv, 2wv+1
  const bool col0 = (lane & 15) == 0;
  const int  hlf  = lane >> 4;
  // Unconditional-B trick: lanes that don't carry column 0 read zeros.
  const __bf16* xsrc = col0 ? (const __bf16*)xb : (const __bf16*)zerov;
  const __bf16* zsrc = col0 ? (const __bf16*)zb : (const __bf16*)zerov;

  // init states and LDS-resident constants (fused biases)
  h0[t] = 0.f;
#pragma unroll
  for (int d = 0; d < D1c; ++d) h1[d][t] = 0.f;
#pragma unroll
  for (int d = 0; d < D2c; ++d) h2[d][t] = 0.f;
  zerov[t] = (__bf16)0.f;
  bsum0[t] = bih0[t] + bhh0[t];
#pragma unroll
  for (int d = 0; d < D1c; ++d) bsum1[d][t] = bih1[d * Hh + t] + bhh1[d * Hh + t];
#pragma unroll
  for (int d = 0; d < D2c; ++d) bsum2[d][t] = bih2[d * Hh + t] + bhh2[d * Hh + t];
  wi0[t] = wih0[t];
  wlo[t] = wlin[t];
  if (t == 0) predAcc = 0.f;
  const float bl = blin[0];
  float pred = q[0];
  float loss = 0.f;
  __syncthreads();

  for (int j = 1; j < TT; ++j) {
    const float qj = q[j];
    const float e0 = pred - qj;
    loss += e0 * e0;
    const int i1 = j & (D1c - 1);
    const int i2 = j & (D2c - 1);

    // ---------- layer 0: new0 = tanh(Whh0 @ h0 + wih0*q + b) ----------
    xb[t] = (__bf16)h0[t];
    __syncthreads();
    matvec1_wmma(whh0p, xsrc, yv, lane, wv, col0, hlf);
    __syncthreads();
    const float v0 = tanhf(yv[t] + wi0[t] * qj + bsum0[t]);
    h0[t] = v0;
    xb[t] = (__bf16)v0;
    zb[t] = (__bf16)h1[i1][t];
    __syncthreads();

    // ---------- layer 1: new1 = tanh(Wih1@new0 + Whh1@h1[i1] + b) ----------
    matvec2_wmma(wih1p + (size_t)i1 * Hh * Hh, whh1p + (size_t)i1 * Hh * Hh,
                 xsrc, zsrc, yv, lane, wv, col0, hlf);
    __syncthreads();
    const float v1 = tanhf(yv[t] + bsum1[i1][t]);
    h1[i1][t] = v1;
    xb[t] = (__bf16)v1;
    zb[t] = (__bf16)h2[i2][t];
    __syncthreads();

    // ---------- layer 2: new2 = tanh(Wih2@new1 + Whh2@h2[i2] + b) ----------
    matvec2_wmma(wih2p + (size_t)i2 * Hh * Hh, whh2p + (size_t)i2 * Hh * Hh,
                 xsrc, zsrc, yv, lane, wv, col0, hlf);
    __syncthreads();
    const float v2 = tanhf(yv[t] + bsum2[i2][t]);
    h2[i2][t] = v2;
    atomicAdd(&predAcc, wlo[t] * v2);   // ds_add_f32 reduction for the head
    __syncthreads();
    pred = predAcc + bl;                // uniform across all threads
    __syncthreads();
    if (t == 0) predAcc = 0.f;          // ordered before next adds by barriers
  }
  if (t == 0) { out[0] = pred; out[1] = loss; }
}

extern "C" void kernel_launch(void* const* d_in, const int* in_sizes, int n_in,
                              void* d_out, int out_size, void* d_ws,
                              size_t ws_size, hipStream_t stream) {
  const float* q    = (const float*)d_in[0];
  const float* wih0 = (const float*)d_in[1];
  const float* whh0 = (const float*)d_in[2];
  const float* bih0 = (const float*)d_in[3];
  const float* bhh0 = (const float*)d_in[4];
  const float* wih1 = (const float*)d_in[5];
  const float* whh1 = (const float*)d_in[6];
  const float* bih1 = (const float*)d_in[7];
  const float* bhh1 = (const float*)d_in[8];
  const float* wih2 = (const float*)d_in[9];
  const float* whh2 = (const float*)d_in[10];
  const float* bih2 = (const float*)d_in[11];
  const float* bhh2 = (const float*)d_in[12];
  const float* wlin = (const float*)d_in[13];
  const float* blin = (const float*)d_in[14];

  const size_t MM = (size_t)Hh * Hh;          // 1M elements per matrix
  __bf16* whh0p = (__bf16*)d_ws;              // 1 matrix
  __bf16* wih1p = whh0p + MM;                 // 2 matrices
  __bf16* whh1p = wih1p + 2 * MM;             // 2 matrices
  __bf16* wih2p = whh1p + 2 * MM;             // 4 matrices
  __bf16* whh2p = wih2p + 4 * MM;             // 4 matrices  (total 26 MB)

  const int thr = 256;
  pack_w<<<(int)(1 * MM / thr), thr, 0, stream>>>(whh0, whh0p, (int)(1 * MM));
  pack_w<<<(int)(2 * MM / thr), thr, 0, stream>>>(wih1, wih1p, (int)(2 * MM));
  pack_w<<<(int)(2 * MM / thr), thr, 0, stream>>>(whh1, whh1p, (int)(2 * MM));
  pack_w<<<(int)(4 * MM / thr), thr, 0, stream>>>(wih2, wih2p, (int)(4 * MM));
  pack_w<<<(int)(4 * MM / thr), thr, 0, stream>>>(whh2, whh2p, (int)(4 * MM));

  rnn_persistent<<<1, 1024, 0, stream>>>(q, wih0, bih0, bhh0, bih1, bhh1, bih2,
                                         bhh2, wlin, blin, whh0p, wih1p, whh1p,
                                         wih2p, whh2p, (float*)d_out);
}
